// MGCN_17532056502542
// MI455X (gfx1250) — matrix operand order repaired and checked by reading, not compiled, not run on hardware
//
#include <hip/hip_runtime.h>
#include <hip/hip_bf16.h>
#include <math.h>

// ---------------------------------------------------------------------------
// Problem constants (match reference)
// ---------------------------------------------------------------------------
#define N_NODES   50000
#define IN_DIM    64
#define HID       256
#define OUT_DIM   8
#define NUM_REL   4
#define N_LAYERS  3
#define E_PER     400000
#define NUM_GRAPHS 64
#define EPS       1e-5f

typedef __attribute__((ext_vector_type(16))) _Float16 v16h;
typedef __attribute__((ext_vector_type(8)))  _Float16 v8h;
typedef __attribute__((ext_vector_type(8)))  float    v8f;

// ---------------------------------------------------------------------------
// WMMA GEMM v2: C[M,N] = A[M,K] * B[K,N] (+ bias[N]); A,B in f16, C f32.
// Block = 256 threads = 8 waves; block tile = 512(M) x 16(N).
// Each wave computes 64x16 (4 accumulators) sharing one B fragment; the
// 32x16 B tile is cooperatively staged TRANSPOSED into LDS each k-step so a
// lane's 16 B halves are one contiguous 32B LDS read.
// Requires N % 16 == 0, K % 32 == 0. M handled with clamp/guard.
// ---------------------------------------------------------------------------
#define BLK_ROWS 512
#define BST      48   // LDS row stride in halves (96B, 32B-aligned, padded)

__device__ inline v16h load_a_frag(const _Float16* Ap) {
  // element j -> K=kbase+j (contiguous 16)
  v8h lo = *reinterpret_cast<const v8h*>(Ap);
  v8h hi = *reinterpret_cast<const v8h*>(Ap + 8);
  return __builtin_shufflevector(lo, hi, 0,1,2,3,4,5,6,7,8,9,10,11,12,13,14,15);
}

__global__ void wmma_gemm_f16v2(const _Float16* __restrict__ A,
                                const _Float16* __restrict__ B,
                                const float* __restrict__ bias,
                                float* __restrict__ C,
                                int M, int N, int K) {
  __shared__ alignas(32) _Float16 Bsh[16 * BST];   // Bsh[n*BST + k], k in [0,32)

  const int tilesN = N >> 4;
  const int bm  = blockIdx.x / tilesN;
  const int tn  = (blockIdx.x % tilesN) << 4;
  const int t   = threadIdx.x;
  const int wave = t >> 5;
  const int lane = t & 31;
  const int tmw  = bm * BLK_ROWS + wave * 64;      // wave's first row

  const int m   = lane & 15;                       // A row / B,C col within tile
  const int hi  = lane >> 4;                       // lane-group
  const int kba = hi << 4;                         // A: +16 halves for lanes 16-31
  const int kbb = hi << 4;                         // B: +16 halves for lanes 16-31

  // clamped A row pointers for the 4 M-subtiles (all threads stay alive)
  const _Float16* Arow[4];
#pragma unroll
  for (int s = 0; s < 4; ++s) {
    int r = tmw + s * 16 + m;
    if (r > M - 1) r = M - 1;
    Arow[s] = A + (size_t)r * K;
  }

  v8f acc0 = {}, acc1 = {}, acc2 = {}, acc3 = {};

  for (int k0 = 0; k0 < K; k0 += 32) {
    __syncthreads();   // protect previous iteration's Bsh readers
    {
      // stage 32x16 B tile transposed: Bsh[n][k] = B[k0+k][tn+n]
      int j0 = t, j1 = t + 256;                    // 512 halves total
      Bsh[(j0 & 15) * BST + (j0 >> 4)] = B[(size_t)(k0 + (j0 >> 4)) * N + tn + (j0 & 15)];
      Bsh[(j1 & 15) * BST + (j1 >> 4)] = B[(size_t)(k0 + (j1 >> 4)) * N + tn + (j1 & 15)];
    }
    __syncthreads();

    // B fragment: lane's 16 halves are contiguous (k = kbb .. kbb+15)
    const v16h bf = *reinterpret_cast<const v16h*>(&Bsh[m * BST + kbb]);

    acc0 = __builtin_amdgcn_wmma_f32_16x16x32_f16(false, load_a_frag(Arow[0] + k0 + kba),
                                                  false, bf, (short)0, acc0, false, false);
    acc1 = __builtin_amdgcn_wmma_f32_16x16x32_f16(false, load_a_frag(Arow[1] + k0 + kba),
                                                  false, bf, (short)0, acc1, false, false);
    acc2 = __builtin_amdgcn_wmma_f32_16x16x32_f16(false, load_a_frag(Arow[2] + k0 + kba),
                                                  false, bf, (short)0, acc2, false, false);
    acc3 = __builtin_amdgcn_wmma_f32_16x16x32_f16(false, load_a_frag(Arow[3] + k0 + kba),
                                                  false, bf, (short)0, acc3, false, false);
  }

  // C/D layout: VGPR v -> row (subtile + v + 8*hi), col tn + m
  const int col = tn + m;
  const float bv = bias ? bias[col] : 0.0f;
#define STORE_SUB(s, accv)                                                    \
  {                                                                           \
    const int rbase = tmw + (s) * 16 + (hi << 3);                             \
    _Pragma("unroll")                                                         \
    for (int v = 0; v < 8; ++v) {                                             \
      int r = rbase + v;                                                      \
      if (r < M) C[(size_t)r * N + col] = (accv)[v] + bv;                     \
    }                                                                         \
  }
  STORE_SUB(0, acc0) STORE_SUB(1, acc1) STORE_SUB(2, acc2) STORE_SUB(3, acc3)
#undef STORE_SUB
}

// ---------------------------------------------------------------------------
// Elementwise / graph kernels
// ---------------------------------------------------------------------------
__global__ void cvt_f16(const float* __restrict__ X, _Float16* __restrict__ Y, int n) {
  int i = blockIdx.x * blockDim.x + threadIdx.x;
  if (i < n) Y[i] = (_Float16)X[i];
}

__global__ void fill_f32(float* p, float v, int n) {
  int i = blockIdx.x * blockDim.x + threadIdx.x;
  if (i < n) p[i] = v;
}

__global__ void edge_deg(const int* __restrict__ col, const float* __restrict__ ew,
                         float* __restrict__ deg, int E) {
  int e = blockIdx.x * blockDim.x + threadIdx.x;
  if (e < E) atomicAdd(&deg[col[e]], ew[e]);
}

__global__ void deg_to_dinv(float* d, int n) {   // in place: d = rsqrt(d + 1)
  int i = blockIdx.x * blockDim.x + threadIdx.x;
  if (i < n) d[i] = rsqrtf(d[i] + 1.0f);
}

__global__ void edge_norm(const int* __restrict__ row, const int* __restrict__ col,
                          const float* __restrict__ ew, const float* __restrict__ dinv,
                          float* __restrict__ norm, int E) {
  int e = blockIdx.x * blockDim.x + threadIdx.x;
  if (e < E) norm[e] = dinv[row[e]] * ew[e] * dinv[col[e]];
}

// out[col] += norm * Hr[row] — one wave per edge, lanes stride 256 channels
__global__ void scatter_edges(const float* __restrict__ Hr,
                              const int* __restrict__ row, const int* __restrict__ col,
                              const float* __restrict__ norm,
                              float* __restrict__ out, int E) {
  int e = (blockIdx.x * blockDim.x + threadIdx.x) >> 5;
  if (e >= E) return;
  int lane = threadIdx.x & 31;
  const int r = row[e], c = col[e];
  const float nv = norm[e];
  const float* src = Hr  + (size_t)r * HID;
  float*       dst = out + (size_t)c * HID;
  __builtin_prefetch(src, 0, 1);   // gfx1250 global_prefetch_b8 on gather stream
#pragma unroll
  for (int j = 0; j < HID; j += 32)
    atomicAdd(&dst[j + lane], nv * src[j + lane]);
}

// out[i,:] += Hr[i,:] * dinv[i]^2 + bias[:]
__global__ void selfloop_bias(float* __restrict__ out, const float* __restrict__ Hr,
                              const float* __restrict__ dinv,
                              const float* __restrict__ bias) {
  int i = blockIdx.x, c = threadIdx.x;
  float di = dinv[i];
  size_t idx = (size_t)i * HID + c;
  out[idx] += Hr[idx] * di * di + bias[c];
}

__global__ void bn_stats(const float* __restrict__ X, float* __restrict__ sums,
                         int rows, int C) {
  int c = threadIdx.x;
  int rpb = (rows + gridDim.x - 1) / gridDim.x;
  int r0 = blockIdx.x * rpb;
  int r1 = r0 + rpb; if (r1 > rows) r1 = rows;
  float s = 0.0f, ss = 0.0f;
  for (int r = r0; r < r1; ++r) {
    float v = X[(size_t)r * C + c];
    s += v; ss += v * v;
  }
  if (r0 < r1) { atomicAdd(&sums[c], s); atomicAdd(&sums[C + c], ss); }
}

// BN(train-stats) + ReLU; optionally emits an f16 copy for the next GEMM's A.
__global__ void bn_relu(const float* __restrict__ X, const float* __restrict__ sums,
                        const float* __restrict__ g, const float* __restrict__ b,
                        float* __restrict__ Y, _Float16* __restrict__ Y16,
                        int rows, int C) {
  int idx = blockIdx.x * blockDim.x + threadIdx.x;
  if (idx >= rows * C) return;
  int c = idx % C;
  float m   = sums[c] / (float)rows;
  float var = sums[C + c] / (float)rows - m * m;
  float y = fmaxf(g[c] * (X[idx] - m) * rsqrtf(var + EPS) + b[c], 0.0f);
  Y[idx] = y;
  if (Y16) Y16[idx] = (_Float16)y;
}

__global__ void pool_add(const float* __restrict__ H, const int* __restrict__ batch,
                         float* __restrict__ G) {
  int i = blockIdx.x, c = threadIdx.x;
  atomicAdd(&G[(size_t)batch[i] * HID + c], H[(size_t)i * HID + c]);
}

// out[64,8] = G[64,32] @ W[32,8] + b
__global__ void final_fc(const float* __restrict__ G, const float* __restrict__ W,
                         const float* __restrict__ b, float* __restrict__ out) {
  int gi = blockIdx.x, o = threadIdx.x;
  float s = b[o];
#pragma unroll
  for (int k = 0; k < 32; ++k) s += G[gi * 32 + k] * W[k * 8 + o];
  out[gi * 8 + o] = s;
}

// ---------------------------------------------------------------------------
// Host orchestration
// ---------------------------------------------------------------------------
static inline int cdiv(long long a, long long b) { return (int)((a + b - 1) / b); }

extern "C" void kernel_launch(void* const* d_in, const int* in_sizes, int n_in,
                              void* d_out, int out_size, void* d_ws, size_t ws_size,
                              hipStream_t stream) {
  const float* x     = (const float*)d_in[0];
  const int*   ei    = (const int*)  d_in[1];
  const float* ew    = (const float*)d_in[2];
  const int*   batch = (const int*)  d_in[3];
  const float* pre_w  = (const float*)d_in[4];
  const float* pre_b  = (const float*)d_in[5];
  const float* pre_g  = (const float*)d_in[6];
  const float* pre_be = (const float*)d_in[7];
  const float* conv_w = (const float*)d_in[8];
  const float* conv_b = (const float*)d_in[9];
  const float* skip_w = (const float*)d_in[10];
  const float* skip_b = (const float*)d_in[11];
  const float* bn_g   = (const float*)d_in[12];
  const float* bn_b   = (const float*)d_in[13];
  const float* p1_w   = (const float*)d_in[14];
  const float* p1_b   = (const float*)d_in[15];
  const float* p1_g   = (const float*)d_in[16];
  const float* p1_be  = (const float*)d_in[17];
  const float* p2_w   = (const float*)d_in[18];
  const float* p2_b   = (const float*)d_in[19];
  const float* p2_g   = (const float*)d_in[20];
  const float* p2_be  = (const float*)d_in[21];
  const float* p3_w   = (const float*)d_in[22];
  const float* p3_b   = (const float*)d_in[23];

  // ---- workspace carve-up (256B aligned)
  char* ws = (char*)d_ws;
  size_t off = 0;
  auto carve = [&](size_t bytes) -> void* {
    void* p = (void*)(ws + off);
    off += (bytes + 255) & ~(size_t)255;
    return p;
  };
  const size_t NHID = (size_t)N_NODES * HID * sizeof(float);
  float*    h     = (float*)carve(NHID);
  float*    out   = (float*)carve(NHID);
  float*    Hr    = (float*)carve(NHID);
  float*    dinv  = (float*)carve((size_t)NUM_REL * N_NODES * sizeof(float));
  float*    norm  = (float*)carve((size_t)NUM_REL * E_PER * sizeof(float));
  float*    stats = (float*)carve(2 * HID * sizeof(float));
  float*    G     = (float*)carve((size_t)NUM_GRAPHS * HID * sizeof(float));
  float*    P1    = (float*)carve((size_t)NUM_GRAPHS * HID * sizeof(float));
  float*    P2    = (float*)carve((size_t)NUM_GRAPHS * 32 * sizeof(float));
  _Float16* h16   = (_Float16*)carve((size_t)N_NODES * HID * 2);
  _Float16* x16   = (_Float16*)carve((size_t)N_NODES * IN_DIM * 2);
  _Float16* preW16  = (_Float16*)carve((size_t)IN_DIM * HID * 2);
  _Float16* convW16 = (_Float16*)carve((size_t)N_LAYERS * NUM_REL * HID * HID * 2);
  _Float16* skipW16 = (_Float16*)carve((size_t)N_LAYERS * HID * HID * 2);
  _Float16* p1W16   = (_Float16*)carve((size_t)HID * HID * 2);
  _Float16* p2W16   = (_Float16*)carve((size_t)HID * 32 * 2);
  _Float16* G16     = (_Float16*)carve((size_t)NUM_GRAPHS * HID * 2);
  _Float16* P1h     = (_Float16*)carve((size_t)NUM_GRAPHS * HID * 2);
  (void)ws_size; (void)n_in; (void)in_sizes; (void)out_size;

  const int TB = 256;
  auto cvt = [&](const float* src, _Float16* dst, int n) {
    cvt_f16<<<cdiv(n, TB), TB, 0, stream>>>(src, dst, n);
  };
  auto gemm = [&](const _Float16* A, const _Float16* B, const float* bias, float* C,
                  int M, int N, int K) {
    wmma_gemm_f16v2<<<cdiv(M, BLK_ROWS) * (N / 16), TB, 0, stream>>>(A, B, bias, C, M, N, K);
  };

  // ---- one-time f16 conversions (weights + input features)
  cvt(x, x16, N_NODES * IN_DIM);
  cvt(pre_w,  preW16,  IN_DIM * HID);
  cvt(conv_w, convW16, N_LAYERS * NUM_REL * HID * HID);
  cvt(skip_w, skipW16, N_LAYERS * HID * HID);
  cvt(p1_w,   p1W16,   HID * HID);
  cvt(p2_w,   p2W16,   HID * 32);

  // ---- precompute: deg -> dinv -> per-edge norm (layer-invariant)
  fill_f32<<<cdiv(NUM_REL * N_NODES, TB), TB, 0, stream>>>(dinv, 0.0f, NUM_REL * N_NODES);
  for (int r = 0; r < NUM_REL; ++r) {
    const int* col_r = ei + ((size_t)r * 2 + 1) * E_PER;
    edge_deg<<<cdiv(E_PER, TB), TB, 0, stream>>>(col_r, ew + (size_t)r * E_PER,
                                                 dinv + (size_t)r * N_NODES, E_PER);
  }
  deg_to_dinv<<<cdiv(NUM_REL * N_NODES, TB), TB, 0, stream>>>(dinv, NUM_REL * N_NODES);
  for (int r = 0; r < NUM_REL; ++r) {
    const int* row_r = ei + ((size_t)r * 2 + 0) * E_PER;
    const int* col_r = ei + ((size_t)r * 2 + 1) * E_PER;
    edge_norm<<<cdiv(E_PER, TB), TB, 0, stream>>>(row_r, col_r, ew + (size_t)r * E_PER,
                                                  dinv + (size_t)r * N_NODES,
                                                  norm + (size_t)r * E_PER, E_PER);
  }

  // ---- pre_mp: Linear -> BN -> ReLU (emit f16 activations)
  gemm(x16, preW16, pre_b, out, N_NODES, HID, IN_DIM);
  fill_f32<<<cdiv(2 * HID, TB), TB, 0, stream>>>(stats, 0.0f, 2 * HID);
  bn_stats<<<512, HID, 0, stream>>>(out, stats, N_NODES, HID);
  bn_relu<<<cdiv((long long)N_NODES * HID, TB), TB, 0, stream>>>(
      out, stats, pre_g, pre_be, h, h16, N_NODES, HID);

  // ---- message-passing layers
  for (int l = 0; l < N_LAYERS; ++l) {
    gemm(h16, skipW16 + (size_t)l * HID * HID, skip_b + (size_t)l * HID, out,
         N_NODES, HID, HID);
    for (int r = 0; r < NUM_REL; ++r) {
      const int lr = l * NUM_REL + r;
      const int* row_r = ei + ((size_t)r * 2 + 0) * E_PER;
      const int* col_r = ei + ((size_t)r * 2 + 1) * E_PER;
      gemm(h16, convW16 + (size_t)lr * HID * HID, nullptr, Hr, N_NODES, HID, HID);
      selfloop_bias<<<N_NODES, HID, 0, stream>>>(out, Hr, dinv + (size_t)r * N_NODES,
                                                 conv_b + (size_t)lr * HID);
      scatter_edges<<<cdiv((long long)E_PER * 32, TB), TB, 0, stream>>>(
          Hr, row_r, col_r, norm + (size_t)r * E_PER, out, E_PER);
    }
    fill_f32<<<cdiv(2 * HID, TB), TB, 0, stream>>>(stats, 0.0f, 2 * HID);
    bn_stats<<<512, HID, 0, stream>>>(out, stats, N_NODES, HID);
    bn_relu<<<cdiv((long long)N_NODES * HID, TB), TB, 0, stream>>>(
        out, stats, bn_g + (size_t)l * HID, bn_b + (size_t)l * HID, h, h16, N_NODES, HID);
  }

  // ---- global_add_pool
  fill_f32<<<cdiv(NUM_GRAPHS * HID, TB), TB, 0, stream>>>(G, 0.0f, NUM_GRAPHS * HID);
  pool_add<<<N_NODES, HID, 0, stream>>>(h, batch, G);

  // ---- post_mp
  cvt(G, G16, NUM_GRAPHS * HID);
  gemm(G16, p1W16, p1_b, P1, NUM_GRAPHS, HID, HID);
  fill_f32<<<cdiv(2 * HID, TB), TB, 0, stream>>>(stats, 0.0f, 2 * HID);
  bn_stats<<<NUM_GRAPHS, HID, 0, stream>>>(P1, stats, NUM_GRAPHS, HID);
  bn_relu<<<cdiv(NUM_GRAPHS * HID, TB), TB, 0, stream>>>(
      P1, stats, p1_g, p1_be, P1, P1h, NUM_GRAPHS, HID);

  gemm(P1h, p2W16, p2_b, P2, NUM_GRAPHS, 32, HID);
  fill_f32<<<1, 64, 0, stream>>>(stats, 0.0f, 64);
  bn_stats<<<NUM_GRAPHS, 32, 0, stream>>>(P2, stats, NUM_GRAPHS, 32);
  bn_relu<<<cdiv(NUM_GRAPHS * 32, TB), TB, 0, stream>>>(
      P2, stats, p2_g, p2_be, P2, nullptr, NUM_GRAPHS, 32);

  final_fc<<<NUM_GRAPHS, OUT_DIM, 0, stream>>>(P2, p3_w, p3_b, (float*)d_out);
}
